// DeformableLocalAttention_14826227106561
// MI455X (gfx1250) — matrix operand order repaired and checked by reading, not compiled, not run on hardware
//
#include <hip/hip_runtime.h>
#include <hip/hip_bf16.h>
#include <math.h>

typedef __bf16 bf16_t;
typedef __attribute__((ext_vector_type(16))) __bf16 v16bf;
typedef __attribute__((ext_vector_type(8)))  float  v8f;

constexpr int Bb = 2, Nn = 1024, Cc = 384, Kk = 10, Gg = 2, GD = 192;
constexpr float ATT_SCALE = 0.125f;   // 64^-0.5
constexpr float LN_EPS = 1e-5f;

__device__ __forceinline__ bf16_t f2bf(float f) { return (bf16_t)f; }

// Low 32 bits of a generic LDS pointer are the LDS byte offset (ISA 10.2).
__device__ __forceinline__ uint32_t lds_lo(const void* p) {
  return (uint32_t)(uintptr_t)p;
}
// CDNA5 async global->LDS copy, tracked by ASYNCcnt (ISA 10.x / 08_async_tensor).
__device__ __forceinline__ void async_copy_b64(uint32_t lds_addr, const bf16_t* gptr) {
  asm volatile("global_load_async_to_lds_b64 %0, %1, off"
               :: "v"(lds_addr), "v"((uint64_t)(uintptr_t)gptr)
               : "memory");
}
__device__ __forceinline__ void wait_async0() {
  asm volatile("s_wait_asynccnt 0x0" ::: "memory");
}
// pack 4 f32 -> 4 bf16 and store 8B to LDS
__device__ __forceinline__ void stage4(bf16_t* dst, float4 v) {
  union { bf16_t h[4]; uint2 u2; } pk;
  pk.h[0] = f2bf(v.x); pk.h[1] = f2bf(v.y); pk.h[2] = f2bf(v.z); pk.h[3] = f2bf(v.w);
  *(uint2*)dst = pk.u2;
}

// ---- WMMA fragment loaders (layouts per CDNA5 ISA 7.12.2, wave32) ----
__device__ __forceinline__ v16bf load_a_frag(const bf16_t* As, int stride, int mrow, int hi) {
  union { v16bf v; uint32_t u[8]; } f;
  const bf16_t* row = As + mrow * stride;
#pragma unroll
  for (int r = 0; r < 8; ++r) {
    int k0 = (r < 4) ? ((hi ? 8 : 0) + 2 * r) : ((hi ? 24 : 16) + 2 * (r - 4));
    f.u[r] = *(const uint32_t*)(row + k0);
  }
  return f.v;
}
__device__ __forceinline__ v16bf load_b_frag(const bf16_t* Bs, int stride, int nrow, int hi) {
  union { v16bf v; uint32_t u[8]; } f;
  const bf16_t* row = Bs + nrow * stride;
#pragma unroll
  for (int r = 0; r < 8; ++r) {
    int k0 = (r < 4) ? ((hi ? 8 : 0) + 2 * r) : ((hi ? 24 : 16) + 2 * (r - 4));
    f.u[r] = *(const uint32_t*)(row + k0);
  }
  return f.v;
}
__device__ __forceinline__ v8f wmma_bf16(v16bf a, v16bf b, v8f c) {
  return __builtin_amdgcn_wmma_f32_16x16x32_bf16(false, a, false, b, (short)0, c, false, false);
}

// ---- 0) weight convert: WT[n][k] = bf16(W[k][n]), 384x384 ----
__global__ __launch_bounds__(256) void wconv_kernel(const float* __restrict__ W,
                                                    bf16_t* __restrict__ WT) {
  int t = blockIdx.x * 256 + threadIdx.x;      // n*384+k
  int n = t / 384, k = t % 384;
  WT[t] = f2bf(W[k * 384 + n]);
}

// ---- 1) kNN top-10 over pos (brute force, LDS-tiled) ----
__global__ __launch_bounds__(256) void knn_kernel(const float* __restrict__ pos,
                                                  int* __restrict__ idx) {
  int q = blockIdx.x * 256 + threadIdx.x;      // 0..2047, block never straddles batch
  int b = q >> 10;
  float qx = pos[q * 3 + 0], qy = pos[q * 3 + 1], qz = pos[q * 3 + 2];
  float dr[Kk]; int ir[Kk];
#pragma unroll
  for (int i = 0; i < Kk; ++i) { dr[i] = 3.0e38f; ir[i] = 0; }
  __shared__ float P[256][3];
  for (int tile = 0; tile < 4; ++tile) {
    __syncthreads();
    int t = threadIdx.x, j = tile * 256 + t;
    P[t][0] = pos[((size_t)b * Nn + j) * 3 + 0];
    P[t][1] = pos[((size_t)b * Nn + j) * 3 + 1];
    P[t][2] = pos[((size_t)b * Nn + j) * 3 + 2];
    __syncthreads();
    for (int jj = 0; jj < 256; ++jj) {
      float dx = qx - P[jj][0], dy = qy - P[jj][1], dz = qz - P[jj][2];
      float d = dx * dx + dy * dy + dz * dz;
      if (d < dr[Kk - 1]) {
        int p = Kk - 1;
        while (p > 0 && dr[p - 1] > d) { dr[p] = dr[p - 1]; ir[p] = ir[p - 1]; --p; }
        dr[p] = d; ir[p] = tile * 256 + jj;
      }
    }
  }
#pragma unroll
  for (int i = 0; i < Kk; ++i) idx[(size_t)q * Kk + i] = ir[i];
}

// ---- generic WMMA GEMM: out[M,384] = A[M,384] @ W + bias ; 64x128 block tile ----
__global__ __launch_bounds__(128) void gemm_bf16w_kernel(const float* __restrict__ A,
                                                         const bf16_t* __restrict__ WT,
                                                         const float* __restrict__ bias,
                                                         float* __restrict__ out, int M) {
  __shared__ bf16_t As[64][40];
  __shared__ bf16_t Bs[128][40];
  const int tid = threadIdx.x, wave = tid >> 5, lane = tid & 31;
  const int hi = lane >> 4, ln = lane & 15;
  const int r0 = blockIdx.x * 64, c0 = blockIdx.y * 128;
  v8f acc[8] = {};
  for (int kt = 0; kt < 12; ++kt) {
    __syncthreads();
#pragma unroll
    for (int i = 0; i < 4; ++i) {                 // A: 64x32 f32 -> bf16, float4 granularity
      int cid = i * 128 + tid;
      int rl = cid >> 3, kk = (cid & 7) * 4;
      float4 av = *(const float4*)(A + (size_t)(r0 + rl) * 384 + kt * 32 + kk);
      stage4(&As[rl][kk], av);
    }
#pragma unroll
    for (int i = 0; i < 8; ++i) {                 // B: 128x32 bf16, async global->LDS b64
      int cid = i * 128 + tid;
      int rl = cid >> 3, kk = (cid & 7) * 4;
      async_copy_b64(lds_lo(&Bs[rl][kk]), WT + (size_t)(c0 + rl) * 384 + kt * 32 + kk);
    }
    wait_async0();
    __syncthreads();
    v16bf a = load_a_frag(&As[0][0], 40, wave * 16 + ln, hi);
#pragma unroll
    for (int ct = 0; ct < 8; ++ct) {
      v16bf bfr = load_b_frag(&Bs[0][0], 40, ct * 16 + ln, hi);
      acc[ct] = wmma_bf16(a, bfr, acc[ct]);
    }
  }
  int mb = r0 + wave * 16;
#pragma unroll
  for (int ct = 0; ct < 8; ++ct) {
    int col = c0 + ct * 16 + ln;
    float bv = bias ? bias[col] : 0.0f;
#pragma unroll
    for (int r = 0; r < 8; ++r)
      out[(size_t)(mb + r + 8 * hi) * 384 + col] = acc[ct][r] + bv;
  }
}

// ---- fused gather + WMMA GEMM: h1 = shift_feat @ W1 (rows gathered from v_off / q) ----
__global__ __launch_bounds__(128) void gemm_shift_kernel(const float* __restrict__ voff,
                                                         const float* __restrict__ q,
                                                         const int* __restrict__ idx,
                                                         const bf16_t* __restrict__ W1T,
                                                         float* __restrict__ h1) {
  __shared__ bf16_t As[64][40];
  __shared__ bf16_t Bs[128][40];
  __shared__ int base1[64], base2[64];
  const int tid = threadIdx.x, wave = tid >> 5, lane = tid & 31;
  const int hi = lane >> 4, ln = lane & 15;
  const int r0 = blockIdx.x * 64, c0 = blockIdx.y * 128;
  if (tid < 64) {
    int r = r0 + tid;
    int k = r % Kk, n = (r / Kk) % Nn, g = (r / (Kk * Nn)) % Gg, b = r / (Kk * Nn * Gg);
    int j = idx[((size_t)b * Nn + n) * Kk + k];
    base1[tid] = (b * Nn + j) * 384 + g * GD;   // v_off gather row, group channels
    base2[tid] = (b * Nn + n) * 384 + g * GD;   // q row, group channels
  }
  v8f acc[8] = {};
  for (int kt = 0; kt < 12; ++kt) {
    __syncthreads();
    const float* src = (kt < 6) ? voff : q;
    const int* base = (kt < 6) ? base1 : base2;
    int cofs = (kt < 6) ? kt * 32 : kt * 32 - 192;
#pragma unroll
    for (int i = 0; i < 4; ++i) {
      int cid = i * 128 + tid;
      int rl = cid >> 3, kk = (cid & 7) * 4;
      float4 av = *(const float4*)(src + (size_t)base[rl] + cofs + kk);
      stage4(&As[rl][kk], av);
    }
#pragma unroll
    for (int i = 0; i < 8; ++i) {
      int cid = i * 128 + tid;
      int rl = cid >> 3, kk = (cid & 7) * 4;
      async_copy_b64(lds_lo(&Bs[rl][kk]), W1T + (size_t)(c0 + rl) * 384 + kt * 32 + kk);
    }
    wait_async0();
    __syncthreads();
    v16bf a = load_a_frag(&As[0][0], 40, wave * 16 + ln, hi);
#pragma unroll
    for (int ct = 0; ct < 8; ++ct) {
      v16bf bfr = load_b_frag(&Bs[0][0], 40, ct * 16 + ln, hi);
      acc[ct] = wmma_bf16(a, bfr, acc[ct]);
    }
  }
  int mb = r0 + wave * 16;
#pragma unroll
  for (int ct = 0; ct < 8; ++ct) {
    int col = c0 + ct * 16 + ln;
#pragma unroll
    for (int r = 0; r < 8; ++r)
      h1[(size_t)(mb + r + 8 * hi) * 384 + col] = acc[ct][r];
  }
}

// ---- LN + GELU + @W2 + tanh + pos add -> shift_pos ----
__global__ __launch_bounds__(128) void ln_gelu_w2_kernel(const float* __restrict__ h1,
                                                         const float* __restrict__ b1,
                                                         const float* __restrict__ ln_g,
                                                         const float* __restrict__ ln_b,
                                                         const float* __restrict__ W2,
                                                         const float* __restrict__ pos,
                                                         const int* __restrict__ idx,
                                                         float* __restrict__ shift_pos) {
  __shared__ float red[128], red2[128];
  int r = blockIdx.x, t = threadIdx.x;
  float v[3], s = 0.f, s2 = 0.f;
#pragma unroll
  for (int i = 0; i < 3; ++i) {
    int c = t + i * 128;
    float xv = h1[(size_t)r * 384 + c] + b1[c];
    v[i] = xv; s += xv; s2 += xv * xv;
  }
  red[t] = s; red2[t] = s2; __syncthreads();
  for (int o2 = 64; o2 > 0; o2 >>= 1) {
    if (t < o2) { red[t] += red[t + o2]; red2[t] += red2[t + o2]; }
    __syncthreads();
  }
  float mu = red[0] * (1.0f / 384.0f);
  float var = red2[0] * (1.0f / 384.0f) - mu * mu;
  float rstd = rsqrtf(var + LN_EPS);
  __syncthreads();
  float d[3] = {0.f, 0.f, 0.f};
#pragma unroll
  for (int i = 0; i < 3; ++i) {
    int c = t + i * 128;
    float y = (v[i] - mu) * rstd * ln_g[c] + ln_b[c];
    float ge = 0.5f * y * (1.0f + erff(y * 0.70710678f));   // exact GELU
#pragma unroll
    for (int j = 0; j < 3; ++j) d[j] += ge * W2[c * 3 + j];
  }
  float dres[3];
#pragma unroll
  for (int j = 0; j < 3; ++j) {
    red[t] = d[j]; __syncthreads();
    for (int o2 = 64; o2 > 0; o2 >>= 1) {
      if (t < o2) red[t] += red[t + o2];
      __syncthreads();
    }
    if (t == 0) dres[j] = red[0];
    __syncthreads();
  }
  if (t == 0) {
    int k = r % Kk, n = (r / Kk) % Nn, b = r / (Kk * Nn * Gg);
    int j = idx[((size_t)b * Nn + n) * Kk + k];
#pragma unroll
    for (int c = 0; c < 3; ++c)
      shift_pos[(size_t)r * 3 + c] = pos[((size_t)b * Nn + j) * 3 + c] + tanhf(dres[c]);
  }
}

// ---- three_nn: top-3 neighbors + interpolation weights ----
__global__ __launch_bounds__(256) void three_nn_kernel(const float* __restrict__ shift_pos,
                                                       const float* __restrict__ pos,
                                                       int* __restrict__ idx3,
                                                       float* __restrict__ w3) {
  int p = blockIdx.x * 256 + threadIdx.x;         // 0..40959; 20480 per batch -> no straddle
  int b = p / (Gg * Nn * Kk);
  float px = shift_pos[p * 3], py = shift_pos[p * 3 + 1], pz = shift_pos[p * 3 + 2];
  float d0 = 3e38f, d1 = 3e38f, d2 = 3e38f;
  int i0 = 0, i1 = 0, i2 = 0;
  __shared__ float P[256][3];
  for (int tile = 0; tile < 4; ++tile) {
    __syncthreads();
    int t = threadIdx.x, j = tile * 256 + t;
    P[t][0] = pos[((size_t)b * Nn + j) * 3 + 0];
    P[t][1] = pos[((size_t)b * Nn + j) * 3 + 1];
    P[t][2] = pos[((size_t)b * Nn + j) * 3 + 2];
    __syncthreads();
    for (int jj = 0; jj < 256; ++jj) {
      float dx = px - P[jj][0], dy = py - P[jj][1], dz = pz - P[jj][2];
      float d = dx * dx + dy * dy + dz * dz;
      int g = tile * 256 + jj;
      if (d < d2) {
        if (d < d0)      { d2 = d1; i2 = i1; d1 = d0; i1 = i0; d0 = d; i0 = g; }
        else if (d < d1) { d2 = d1; i2 = i1; d1 = d; i1 = g; }
        else             { d2 = d; i2 = g; }
      }
    }
  }
  float dd[3] = {d0, d1, d2}; int ii[3] = {i0, i1, i2};
  float rc[3], rs = 0.f;
#pragma unroll
  for (int j = 0; j < 3; ++j) { rc[j] = 1.0f / (sqrtf(fmaxf(dd[j], 1e-12f)) + 1e-8f); rs += rc[j]; }
  float inv = 1.0f / rs;
#pragma unroll
  for (int j = 0; j < 3; ++j) { idx3[(size_t)p * 3 + j] = ii[j]; w3[(size_t)p * 3 + j] = rc[j] * inv; }
}

// ---- weighted 3-NN interpolation -> interp[B,N,K,C] ----
__global__ __launch_bounds__(192) void interp_kernel(const float* __restrict__ x,
                                                     const int* __restrict__ idx3,
                                                     const float* __restrict__ w3,
                                                     float* __restrict__ interp) {
  int p = blockIdx.x, cc = threadIdx.x;           // p over B*G*N*K, cc over GD
  int k = p % Kk, n = (p / Kk) % Nn, g = (p / (Kk * Nn)) % Gg, b = p / (Kk * Nn * Gg);
  float acc = 0.f;
#pragma unroll
  for (int j = 0; j < 3; ++j)
    acc += w3[(size_t)p * 3 + j] *
           x[((size_t)b * Nn + idx3[(size_t)p * 3 + j]) * 384 + g * GD + cc];
  interp[(((size_t)b * Nn + n) * Kk + k) * 384 + g * GD + cc] = acc;
}

// ---- tiny local attention (K=10, H=6) + maxpool over K ----
__global__ __launch_bounds__(128) void attn_kernel(const float* __restrict__ q,
                                                   const int* __restrict__ idx,
                                                   const float* __restrict__ kf,
                                                   const float* __restrict__ vf,
                                                   float* __restrict__ pooled) {
  __shared__ float qL[Kk][384], kL[Kk][384], vL[Kk][384], oL[Kk][384];
  int bn = blockIdx.x, t = threadIdx.x;
  int b = bn >> 10;
  for (int e = t; e < Kk * 384; e += 128) {
    int kk = e / 384, c = e % 384;
    int j = idx[(size_t)bn * Kk + kk];
    qL[kk][c] = q[((size_t)b * Nn + j) * 384 + c];
    kL[kk][c] = kf[((size_t)bn * Kk + kk) * 384 + c];
    vL[kk][c] = vf[((size_t)bn * Kk + kk) * 384 + c];
  }
  __syncthreads();
  if (t < 60) {
    int k1 = t / 6, h = t % 6;
    float s[Kk], mx = -3e38f;
#pragma unroll
    for (int k2 = 0; k2 < Kk; ++k2) {
      float a = 0.f;
      for (int c = 0; c < 64; ++c) a += qL[k1][h * 64 + c] * kL[k2][h * 64 + c];
      s[k2] = a * ATT_SCALE;
      mx = fmaxf(mx, s[k2]);
    }
    float den = 0.f;
#pragma unroll
    for (int k2 = 0; k2 < Kk; ++k2) { s[k2] = expf(s[k2] - mx); den += s[k2]; }
    float inv = 1.0f / den;
    for (int c = 0; c < 64; ++c) {
      float a = 0.f;
#pragma unroll
      for (int k2 = 0; k2 < Kk; ++k2) a += s[k2] * vL[k2][h * 64 + c];
      oL[k1][h * 64 + c] = a * inv;
    }
  }
  __syncthreads();
  for (int c = t; c < 384; c += 128) {
    float m = oL[0][c];
#pragma unroll
    for (int k1 = 1; k1 < Kk; ++k1) m = fmaxf(m, oL[k1][c]);
    pooled[(size_t)bn * 384 + c] = m;
  }
}

extern "C" void kernel_launch(void* const* d_in, const int* in_sizes, int n_in,
                              void* d_out, int out_size, void* d_ws, size_t ws_size,
                              hipStream_t stream) {
  (void)in_sizes; (void)n_in; (void)out_size; (void)ws_size;
  const float* x    = (const float*)d_in[0];
  const float* pos  = (const float*)d_in[1];
  const float* Wq   = (const float*)d_in[2];
  const float* Wk   = (const float*)d_in[3];
  const float* Wv   = (const float*)d_in[4];
  const float* Wvo  = (const float*)d_in[5];
  const float* W1   = (const float*)d_in[6];
  const float* b1   = (const float*)d_in[7];
  const float* ln_g = (const float*)d_in[8];
  const float* ln_b = (const float*)d_in[9];
  const float* W2   = (const float*)d_in[10];
  const float* Wp   = (const float*)d_in[11];
  const float* bp   = (const float*)d_in[12];
  float* out = (float*)d_out;

  char* ws = (char*)d_ws;
  size_t o = 0;
  auto take = [&](size_t bytes) -> char* {
    char* p = ws + o;
    o = (o + bytes + 255) & ~(size_t)255;
    return p;
  };
  const size_t WSZ = (size_t)384 * 384;
  bf16_t* wtq = (bf16_t*)take(WSZ * 2);
  bf16_t* wtk = (bf16_t*)take(WSZ * 2);
  bf16_t* wtv = (bf16_t*)take(WSZ * 2);
  bf16_t* wto = (bf16_t*)take(WSZ * 2);
  bf16_t* wt1 = (bf16_t*)take(WSZ * 2);
  bf16_t* wtp = (bf16_t*)take(WSZ * 2);
  float* qb    = (float*)take((size_t)Bb * Nn * 384 * 4);
  float* voffb = (float*)take((size_t)Bb * Nn * 384 * 4);
  int*   idxb  = (int*)take((size_t)Bb * Nn * Kk * 4);
  // big region: h1 [40960,384] f32, later reused as kf|vf [2 x 20480,384] f32
  float* big   = (float*)take((size_t)40960 * 384 * 4);
  float* h1b   = big;
  float* kfb   = big;
  float* vfb   = big + (size_t)20480 * 384;
  float* spb   = (float*)take((size_t)40960 * 3 * 4);
  int*   i3b   = (int*)take((size_t)40960 * 3 * 4);
  float* w3b   = (float*)take((size_t)40960 * 3 * 4);
  float* intb  = (float*)take((size_t)20480 * 384 * 4);
  float* poolb = (float*)take((size_t)Bb * Nn * 384 * 4);

  const int wcB = (384 * 384) / 256;
  wconv_kernel<<<wcB, 256, 0, stream>>>(Wq, wtq);
  wconv_kernel<<<wcB, 256, 0, stream>>>(Wk, wtk);
  wconv_kernel<<<wcB, 256, 0, stream>>>(Wv, wtv);
  wconv_kernel<<<wcB, 256, 0, stream>>>(Wvo, wto);
  wconv_kernel<<<wcB, 256, 0, stream>>>(W1, wt1);
  wconv_kernel<<<wcB, 256, 0, stream>>>(Wp, wtp);

  knn_kernel<<<8, 256, 0, stream>>>(pos, idxb);

  gemm_bf16w_kernel<<<dim3(32, 3), 128, 0, stream>>>(x, wtq, nullptr, qb, 2048);
  gemm_bf16w_kernel<<<dim3(32, 3), 128, 0, stream>>>(x, wto, nullptr, voffb, 2048);

  gemm_shift_kernel<<<dim3(640, 3), 128, 0, stream>>>(voffb, qb, idxb, wt1, h1b);

  ln_gelu_w2_kernel<<<40960, 128, 0, stream>>>(h1b, b1, ln_g, ln_b, W2, pos, idxb, spb);

  three_nn_kernel<<<160, 256, 0, stream>>>(spb, pos, i3b, w3b);

  interp_kernel<<<40960, 192, 0, stream>>>(x, i3b, w3b, intb);

  gemm_bf16w_kernel<<<dim3(320, 3), 128, 0, stream>>>(intb, wtk, nullptr, kfb, 20480);
  gemm_bf16w_kernel<<<dim3(320, 3), 128, 0, stream>>>(intb, wtv, nullptr, vfb, 20480);

  attn_kernel<<<2048, 128, 0, stream>>>(qb, idxb, kfb, vfb, poolb);

  gemm_bf16w_kernel<<<dim3(32, 3), 128, 0, stream>>>(poolb, wtp, bp, out, 2048);
}
